// RecurrentTokenizer_54984171323938
// MI455X (gfx1250) — matrix operand order repaired and checked by reading, not compiled, not run on hardware
//
#include <hip/hip_runtime.h>
#include <hip/hip_bf16.h>

#define N_      32
#define CIN_    1024
#define HW_     1024
#define C_      1024
#define CT_     1024
#define L_      16
#define GROUPS_ 16
#define GS_     64   // group size == head dim

typedef __bf16 bf16;
typedef __attribute__((ext_vector_type(16))) __bf16 v16bf;
typedef __attribute__((ext_vector_type(8)))  __bf16 v8bf;
typedef __attribute__((ext_vector_type(8)))  float  v8f;

union FragAB { v16bf v; v8bf h[2]; bf16 e[16]; };
union FragC  { v8f   v; float e[8]; };

__device__ __forceinline__ v8f wmma_bf16(const FragAB& a, const FragAB& b, v8f c) {
  // D(f32 16x16) = A(bf16 16x32) * B(bf16 32x16) + C
  return __builtin_amdgcn_wmma_f32_16x16x32_bf16(false, a.v, false, b.v, (short)0, c,
                                                 false, false);
}

// ---------------------------------------------------------------------------
// Kernel 1: stream feature -> out (raw passthrough, 134 MB; pure HBM bandwidth)
// ---------------------------------------------------------------------------
__global__ void feature_copy_kernel(const float4* __restrict__ in,
                                    float4* __restrict__ out, int n4) {
  int i = blockIdx.x * blockDim.x + threadIdx.x;
  int stride = gridDim.x * blockDim.x;
  for (; i < n4; i += stride) out[i] = in[i];
}

// ---------------------------------------------------------------------------
// Kernel 2: fused q / key / logits / softmax / value / tok per (n, g) block.
// 128 threads = 4 waves (wave32).
// ---------------------------------------------------------------------------
__global__ __launch_bounds__(128)
void attn_kernel(const float* __restrict__ feature,  // [N][C][HW]
                 const float* __restrict__ token,    // [N][CT][L]
                 const float* __restrict__ Wq,       // [C][CT]
                 const float* __restrict__ bq,       // [C]
                 const float* __restrict__ Wk,       // [G][GS][GS]
                 const float* __restrict__ bk,       // [C]
                 const float* __restrict__ Wv,       // [G][GS][GS]
                 const float* __restrict__ bv,       // [C]
                 float* __restrict__ outTok) {       // [N][C][L] flat
  const int blk   = blockIdx.x;
  const int n     = blk / GROUPS_;
  const int g     = blk % GROUPS_;
  const int tid   = threadIdx.x;
  const int w     = tid >> 5;          // wave id 0..3
  const int lane  = tid & 31;
  const int l16   = lane & 15;
  const int hi    = (lane >> 4) & 1;   // lane >= 16

  // ---- dynamic LDS carve-up -------------------------------------------------
  extern __shared__ __align__(16) char smem[];
  bf16*  sWk   = (bf16*)smem;                        // [64][64]     8192 B
  bf16*  sWv   = sWk + GS_ * GS_;                    // [64][64]     8192 B
  bf16*  sQT   = sWv + GS_ * GS_;                    // [16 l][64 d] 2048 B
  bf16*  sKT   = sQT + L_ * GS_;                     // 4 x [16x][64d] 8192 B
  bf16*  sVT   = sKT + 4 * 16 * GS_;                 // 4 x [16d][32x] 4096 B
  bf16*  sP    = sVT + 4 * 16 * 32;                  // [16 l][1024 x] 32768 B
  float* sCoef = (float*)(sP + L_ * HW_);            // [1024 x][16 l] 65536 B
  float* sRed  = sCoef + HW_ * L_;                   // [8][16]        512 B
  float* sMax  = sRed + 128;                         // [16]
  float* sRcp  = sMax + 16;                          // [16]

  const float* fbase = feature + ((size_t)n * C_ + (size_t)g * GS_) * HW_;
  const float* tbase = token + (size_t)n * CT_ * L_;

  // ---- preload block-diagonal weights (f32 -> bf16) into LDS ---------------
  {
    const float* wkp = Wk + (size_t)g * GS_ * GS_;
    const float* wvp = Wv + (size_t)g * GS_ * GS_;
    for (int i = tid; i < GS_ * GS_; i += 128) {
      sWk[i] = (bf16)wkp[i];
      sWv[i] = (bf16)wvp[i];
    }
  }

  // ---- Phase 1: q tile per wave: rows c0..c0+15, K = CT = 1024 -------------
  {
    const int c0 = g * GS_ + w * 16;
    FragC qacc;
#pragma unroll
    for (int r = 0; r < 8; ++r) qacc.e[r] = bq[c0 + r + hi * 8];
    for (int k0 = 0; k0 < CT_; k0 += 32) {
      FragAB a;  // Wq[16 x 32]
      const float* arow = Wq + (size_t)(c0 + l16) * CT_ + k0 + hi * 16;
#pragma unroll
      for (int i = 0; i < 8; ++i) a.e[i] = (bf16)arow[i];
#pragma unroll
      for (int i = 0; i < 8; ++i) a.e[8 + i] = (bf16)arow[8 + i];
      FragAB b;  // token[32 x 16]
      const float* bcol = tbase + (size_t)(k0 + hi * 16) * L_ + l16;
#pragma unroll
      for (int i = 0; i < 16; ++i) b.e[i] = (bf16)bcol[i * L_];
      qacc.v = wmma_bf16(a, b, qacc.v);
    }
    // store q^T * (1/sqrt(hd)) as bf16: sQT[l][d]
#pragma unroll
    for (int r = 0; r < 8; ++r)
      sQT[l16 * GS_ + w * 16 + r + hi * 8] = (bf16)(qacc.e[r] * 0.125f);
  }
  __syncthreads();

  // ---- hoist constant fragments --------------------------------------------
  FragAB awk[4][2];  // Wk A-frags, all 4 m-tiles x 2 k-steps
#pragma unroll
  for (int mt = 0; mt < 4; ++mt)
#pragma unroll
    for (int ks = 0; ks < 2; ++ks) {
      const int base = (mt * 16 + l16) * GS_ + ks * 32 + hi * 16;
      awk[mt][ks].h[0] = *(const v8bf*)&sWk[base];
      awk[mt][ks].h[1] = *(const v8bf*)&sWk[base + 8];
    }
  FragAB bqf[2];     // q B-frags (K = d)
#pragma unroll
  for (int ks = 0; ks < 2; ++ks)
    bqf[ks].v = *(const v16bf*)&sQT[l16 * GS_ + ks * 32 + hi * 16];
  FragC kbias[4];
#pragma unroll
  for (int mt = 0; mt < 4; ++mt)
#pragma unroll
    for (int r = 0; r < 8; ++r) kbias[mt].e[r] = bk[g * GS_ + mt * 16 + r + hi * 8];

  // ---- Phase 2: key tiles + logits into sCoef -------------------------------
  bf16* kt = sKT + w * 16 * GS_;  // per-wave staging [16x][64d]
  for (int t = w; t < HW_ / 16; t += 4) {
    const int x0 = t * 16;
    FragAB bf[2];  // f[32i x 16x], bf16
#pragma unroll
    for (int ks = 0; ks < 2; ++ks)
#pragma unroll
      for (int i = 0; i < 16; ++i)
        bf[ks].e[i] = (bf16)fbase[(size_t)(ks * 32 + hi * 16 + i) * HW_ + x0 + l16];
#pragma unroll
    for (int mt = 0; mt < 4; ++mt) {
      FragC dk;
      dk.v = kbias[mt].v;
      dk.v = wmma_bf16(awk[mt][0], bf[0], dk.v);
      dk.v = wmma_bf16(awk[mt][1], bf[1], dk.v);
      // transpose-stage key tile: kt[x][d]
#pragma unroll
      for (int r = 0; r < 8; ++r)
        kt[l16 * GS_ + mt * 16 + r + hi * 8] = (bf16)dk.e[r];
    }
    // logits[x, l] = (key^T)(x,d) . q(d,l)
    FragC lg;
#pragma unroll
    for (int r = 0; r < 8; ++r) lg.e[r] = 0.0f;
#pragma unroll
    for (int ks = 0; ks < 2; ++ks) {
      FragAB a;
      const int base = l16 * GS_ + ks * 32 + hi * 16;
      a.h[0] = *(const v8bf*)&kt[base];
      a.h[1] = *(const v8bf*)&kt[base + 8];
      lg.v = wmma_bf16(a, bqf[ks], lg.v);
    }
#pragma unroll
    for (int r = 0; r < 8; ++r)
      sCoef[(x0 + r + hi * 8) * L_ + l16] = lg.e[r];
  }
  __syncthreads();

  // ---- Phase 3: softmax over x (per l column); p stored unnormalized bf16 --
  {
    const int l = tid & 15;
    const int chunk = tid >> 4;  // 8 chunks of 128 rows
    float m = -1e30f;
    for (int x = chunk * 128; x < chunk * 128 + 128; ++x)
      m = fmaxf(m, sCoef[x * L_ + l]);
    sRed[chunk * 16 + l] = m;
    __syncthreads();
    if (tid < 16) {
      float mm = -1e30f;
      for (int c = 0; c < 8; ++c) mm = fmaxf(mm, sRed[c * 16 + tid]);
      sMax[tid] = mm;
    }
    __syncthreads();
    const float M = sMax[l];
    float s = 0.0f;
    for (int x = chunk * 128; x < chunk * 128 + 128; ++x) {
      float p = __expf(sCoef[x * L_ + l] - M);
      s += p;
      sP[l * HW_ + x] = (bf16)p;  // transposed: sP[l][x]
    }
    __syncthreads();  // sRed reuse
    sRed[chunk * 16 + l] = s;
    __syncthreads();
    if (tid < 16) {
      float ss = 0.0f;
      for (int c = 0; c < 8; ++c) ss += sRed[c * 16 + tid];
      sRcp[tid] = 1.0f / ss;
    }
    __syncthreads();
  }

  // ---- Phase 4: tok(d,l) = sum_x value(d,x) * p(x,l); wave w owns d-tile w --
  {
    FragAB awv[2];
#pragma unroll
    for (int ks = 0; ks < 2; ++ks) {
      const int base = (w * 16 + l16) * GS_ + ks * 32 + hi * 16;
      awv[ks].h[0] = *(const v8bf*)&sWv[base];
      awv[ks].h[1] = *(const v8bf*)&sWv[base + 8];
    }
    FragC acc;
#pragma unroll
    for (int r = 0; r < 8; ++r) acc.e[r] = 0.0f;

    bf16* vt = sVT + w * 16 * 32;  // per-wave staging [16d][32x]
    for (int ch = 0; ch < HW_ / 32; ++ch) {
      const int x0 = ch * 32;
#pragma unroll
      for (int sub = 0; sub < 2; ++sub) {
        FragAB bf[2];
#pragma unroll
        for (int ks = 0; ks < 2; ++ks)
#pragma unroll
          for (int i = 0; i < 16; ++i)
            bf[ks].e[i] =
                (bf16)fbase[(size_t)(ks * 32 + hi * 16 + i) * HW_ + x0 + sub * 16 + l16];
        FragC dv;
#pragma unroll
        for (int r = 0; r < 8; ++r) dv.e[r] = 0.0f;  // bias folded into epilogue
        dv.v = wmma_bf16(awv[0], bf[0], dv.v);
        dv.v = wmma_bf16(awv[1], bf[1], dv.v);
        // transpose-stage value tile: vt[d][x]
#pragma unroll
        for (int r = 0; r < 8; ++r)
          vt[(r + hi * 8) * 32 + sub * 16 + l16] = (bf16)dv.e[r];
      }
      FragAB a;  // value as A: [16d x 32x]
      a.h[0] = *(const v8bf*)&vt[l16 * 32 + hi * 16];
      a.h[1] = *(const v8bf*)&vt[l16 * 32 + hi * 16 + 8];
      FragAB b;  // p as B: [32x x 16l], contiguous in sP[l][x]
      b.v = *(const v16bf*)&sP[l16 * HW_ + x0 + hi * 16];
      acc.v = wmma_bf16(a, b, acc.v);
    }

    // epilogue: normalize by softmax sum, add value bias, write [N][C][L] flat
    const float rcp = sRcp[l16];
    float* outp = outTok + (size_t)n * C_ * L_;
#pragma unroll
    for (int r = 0; r < 8; ++r) {
      const int d = w * 16 + r + hi * 8;
      const int c = g * GS_ + d;
      outp[(size_t)c * L_ + l16] = acc.e[r] * rcp + bv[c];
    }
  }
}

// ---------------------------------------------------------------------------
extern "C" void kernel_launch(void* const* d_in, const int* in_sizes, int n_in,
                              void* d_out, int out_size, void* d_ws, size_t ws_size,
                              hipStream_t stream) {
  (void)in_sizes; (void)n_in; (void)out_size; (void)d_ws; (void)ws_size;
  const float* feature = (const float*)d_in[0];
  const float* token   = (const float*)d_in[1];
  const float* Wq      = (const float*)d_in[2];
  const float* bq      = (const float*)d_in[3];
  const float* Wk      = (const float*)d_in[4];
  const float* bk      = (const float*)d_in[5];
  const float* Wv      = (const float*)d_in[6];
  const float* bv      = (const float*)d_in[7];
  float* out = (float*)d_out;

  const size_t featElems = (size_t)N_ * CIN_ * HW_;  // 33,554,432 f32
  feature_copy_kernel<<<4096, 256, 0, stream>>>(
      (const float4*)feature, (float4*)out, (int)(featElems / 4));

  const size_t shmem =
      (size_t)(GS_ * GS_) * 2 * sizeof(bf16)  // sWk + sWv
      + (size_t)L_ * GS_ * sizeof(bf16)       // sQT
      + 4 * 16 * GS_ * sizeof(bf16)           // sKT
      + 4 * 16 * 32 * sizeof(bf16)            // sVT
      + (size_t)L_ * HW_ * sizeof(bf16)       // sP
      + (size_t)HW_ * L_ * sizeof(float)      // sCoef
      + (128 + 16 + 16) * sizeof(float);      // sRed + sMax + sRcp

  attn_kernel<<<N_ * GROUPS_, 128, shmem, stream>>>(
      feature, token, Wq, bq, Wk, bk, Wv, bv, out + featElems);
}
